// SparseConv3d_63496796504735
// MI455X (gfx1250) — matrix operand order repaired and checked by reading, not compiled, not run on hardware
//
#include <hip/hip_runtime.h>
#include <stdint.h>
#include <type_traits>

typedef __attribute__((ext_vector_type(16))) _Float16 v16h;
typedef __attribute__((ext_vector_type(8)))  float    v8f;

#define KMAX 27   // 3x3x3 kernel volume (fixed)

struct Gather16 { float4 f0, f1, f2, f3; };

__device__ __forceinline__ void issue_gather(const float* __restrict__ x,
                                             int idx, int kb, Gather16& g) {
    const float4* p = (const float4*)(x + (long)idx * 32 + kb);
    g.f0 = p[0];
    g.f1 = p[1];
    const float4* q = (const float4*)(x + (long)idx * 32 + 16 + kb);
    g.f2 = q[0];
    g.f3 = q[1];
}

__global__ __launch_bounds__(256)
void sparse_conv3d_wmma(const float* __restrict__ x,              // [N,32] fp32
                        const float* __restrict__ w,              // [K,32,32] fp32
                        const float* __restrict__ bias,           // [32] fp32
                        const int* __restrict__ kidx,             // [K,N] int32
                        const unsigned char* __restrict__ kvalid, // [K,N] bool
                        float* __restrict__ out,                  // [N,32] fp32
                        int N, int KT)
{
    // Weights in f16, pre-swizzled into WMMA B-matrix lane layout:
    // wlds[c][k][lane][j] : halves (2j,2j+1) = W[k][K=(lane>>4)*8+2j+(j>>2)*8(+1)][c*16+(lane&15)]
    __shared__ __align__(32) uint32_t wlds[2][KMAX][32][8];

    const int tot = 2 * KT * 32 * 8;
    for (int d = threadIdx.x; d < tot; d += blockDim.x) {
        int j  = d & 7;
        int ln = (d >> 3) & 31;
        int k  = (d >> 8) % KT;
        int c  = d / (KT * 256);
        int col = c * 16 + (ln & 15);
        int kk  = ((ln >> 4) << 3) + 2 * j + ((j >> 2) << 3);
        const float* wk = w + k * 32 * 32;
        union { _Float16 h[2]; uint32_t u; } cv;
        cv.h[0] = (_Float16)wk[kk * 32 + col];
        cv.h[1] = (_Float16)wk[(kk + 1) * 32 + col];
        wlds[c][k][ln][j] = cv.u;
    }
    __syncthreads();

    const int lane = threadIdx.x & 31;
    const long n0 = ((long)blockIdx.x * (blockDim.x >> 5) + (threadIdx.x >> 5)) * 16;
    if (n0 >= N) return;

    const int  m     = lane & 15;           // A-matrix row owned by this lane
    const int  kb    = (lane >> 4) * 8;     // A-matrix K chunk base (0 or 8)
    const int  col   = lane & 15;           // C/D column owned by this lane
    const long row   = n0 + m;
    const bool rowOK = row < (long)N;
    const long rowC  = rowOK ? row : (long)(N - 1);

    // Bias folded into the initial accumulators (same column for all 8 C-VGPRs)
    const float b0v = bias[col];
    const float b1v = bias[16 + col];
    v8f acc0 = {b0v, b0v, b0v, b0v, b0v, b0v, b0v, b0v};
    v8f acc1 = {b1v, b1v, b1v, b1v, b1v, b1v, b1v, b1v};

    // Generic-lambda body so KT==27 gets a compile-time trip count (unroll/renaming)
    auto body = [&](auto KTC) {
        const int kt = (int)KTC;

        // ---- software pipeline prologue ----
        int          idx_n = __builtin_nontemporal_load(&kidx[rowC]);
        unsigned int v_n   = __builtin_nontemporal_load(&kvalid[rowC]);

        Gather16 cur;
        issue_gather(x, idx_n, kb, cur);
        unsigned int v_use = v_n;

        {
            const long b1 = (long)((1 < kt) ? 1 : 0) * N + rowC;
            idx_n = __builtin_nontemporal_load(&kidx[b1]);
            v_n   = __builtin_nontemporal_load(&kvalid[b1]);
        }

        #pragma unroll 3
        for (int k = 0; k < kt; ++k) {
            const unsigned int v_cur = v_use;

            // gather for tap min(k+1, kt-1) (harmless redundant issue on last iter)
            Gather16 nxt;
            issue_gather(x, idx_n, kb, nxt);
            v_use = v_n;

            // prefetch idx/valid for tap min(k+2, kt-1) — branchless clamp
            const int  kp = (k + 2 < kt) ? (k + 2) : (kt - 1);
            const long b2 = (long)kp * N + rowC;
            idx_n = __builtin_nontemporal_load(&kidx[b2]);
            v_n   = __builtin_nontemporal_load(&kvalid[b2]);

            // convert tap k to f16 in ISA A layout, branchless validity mask
            union { v16h v; uint32_t u[8]; } a;
            a.v[0]  = (_Float16)cur.f0.x; a.v[1]  = (_Float16)cur.f0.y;
            a.v[2]  = (_Float16)cur.f0.z; a.v[3]  = (_Float16)cur.f0.w;
            a.v[4]  = (_Float16)cur.f1.x; a.v[5]  = (_Float16)cur.f1.y;
            a.v[6]  = (_Float16)cur.f1.z; a.v[7]  = (_Float16)cur.f1.w;
            a.v[8]  = (_Float16)cur.f2.x; a.v[9]  = (_Float16)cur.f2.y;
            a.v[10] = (_Float16)cur.f2.z; a.v[11] = (_Float16)cur.f2.w;
            a.v[12] = (_Float16)cur.f3.x; a.v[13] = (_Float16)cur.f3.y;
            a.v[14] = (_Float16)cur.f3.z; a.v[15] = (_Float16)cur.f3.w;

            const uint32_t msk = (rowOK && v_cur != 0) ? 0xFFFFFFFFu : 0u;
            #pragma unroll
            for (int t = 0; t < 8; ++t) a.u[t] &= msk;

            v16h b0 = *(const v16h*)&wlds[0][k][lane][0];
            v16h b1 = *(const v16h*)&wlds[1][k][lane][0];

            acc0 = __builtin_amdgcn_wmma_f32_16x16x32_f16(false, a.v, false, b0,
                                                          (short)0, acc0, false, false);
            acc1 = __builtin_amdgcn_wmma_f32_16x16x32_f16(false, a.v, false, b1,
                                                          (short)0, acc1, false, false);

            cur = nxt;
        }
    };

    if (KT == KMAX) body(std::integral_constant<int, KMAX>{});
    else            body(KT);

    // C/D layout: VGPR r -> M = (lane>>4)*8 + r, N = lane&15 ; NT stores (write-only)
    const int mBase = (lane >> 4) * 8;
    #pragma unroll
    for (int r = 0; r < 8; ++r) {
        const long rr = n0 + mBase + r;
        if (rr < (long)N) {
            __builtin_nontemporal_store(acc0[r], &out[rr * 32 + col]);
            __builtin_nontemporal_store(acc1[r], &out[rr * 32 + 16 + col]);
        }
    }
}

extern "C" void kernel_launch(void* const* d_in, const int* in_sizes, int n_in,
                              void* d_out, int out_size, void* d_ws, size_t ws_size,
                              hipStream_t stream) {
    const float*         x      = (const float*)d_in[0];
    const float*         w      = (const float*)d_in[1];
    const float*         bias   = (const float*)d_in[2];
    const int*           kidx   = (const int*)d_in[3];
    const unsigned char* kvalid = (const unsigned char*)d_in[4];
    float*               out    = (float*)d_out;

    const int N  = in_sizes[0] / 32;           // [N, C_IN=32]
    int       KT = in_sizes[1] / (32 * 32);    // [K, 32, 32]
    if (KT > KMAX) KT = KMAX;

    const int wavesNeeded = (N + 15) / 16;          // 16 output rows per wave
    const int blocks      = (wavesNeeded + 7) / 8;  // 8 waves (256 thr) per block

    sparse_conv3d_wmma<<<blocks, 256, 0, stream>>>(x, w, bias, kidx, kvalid, out, N, KT);
}
